// GRU_39642548142414
// MI455X (gfx1250) — matrix operand order repaired and checked by reading, not compiled, not run on hardware
//
#include <hip/hip_runtime.h>

#define TSTEPS 512
#define BATCH  64
#define DIN    512
#define HID    1024
#define KDIM   1536   // DIN + HID
#define NWG    64
#define NTHR   256

#ifndef HAVE_AMDGCN_TANH
#if defined(__has_builtin)
#if __has_builtin(__builtin_amdgcn_tanhf)
#define HAVE_AMDGCN_TANH 1
#else
#define HAVE_AMDGCN_TANH 0
#endif
#else
#define HAVE_AMDGCN_TANH 0
#endif
#endif

typedef __attribute__((ext_vector_type(16))) __bf16 v16bf;
typedef __attribute__((ext_vector_type(8)))  __bf16 v8bf;
typedef __attribute__((ext_vector_type(8)))  float  v8f;

// round-to-nearest-even fp32 -> bf16 (stored as ushort so host never sees __bf16)
__device__ __forceinline__ unsigned short f2bfu(float f) {
  unsigned u = __builtin_bit_cast(unsigned, f);
  unsigned r = u + 0x7FFFu + ((u >> 16) & 1u);
  return (unsigned short)(r >> 16);
}

// Fast nonlinearities: prefer the CDNA5 v_tanh_f32 TRANS op; else v_exp + v_rcp.
__device__ __forceinline__ float fast_tanh(float v) {
#if HAVE_AMDGCN_TANH
  return __builtin_amdgcn_tanhf(v);
#else
  float e = __expf(-2.0f * v);
  return (1.0f - e) * __builtin_amdgcn_rcpf(1.0f + e);
#endif
}

__device__ __forceinline__ float sigm(float v) {
#if HAVE_AMDGCN_TANH
  return 0.5f + 0.5f * __builtin_amdgcn_tanhf(0.5f * v);
#else
  return __builtin_amdgcn_rcpf(1.0f + __expf(-v));
#endif
}

__device__ __forceinline__ void grid_barrier(unsigned* arrive, unsigned target) {
  __threadfence();               // release this thread's stores to agent scope
  __syncthreads();
  if (threadIdx.x == 0) {
    __hip_atomic_fetch_add(arrive, 1u, __ATOMIC_RELEASE, __HIP_MEMORY_SCOPE_AGENT);
    while (__hip_atomic_load(arrive, __ATOMIC_ACQUIRE, __HIP_MEMORY_SCOPE_AGENT) < target) {
      __builtin_amdgcn_s_sleep(2);
    }
  }
  __syncthreads();
  __threadfence();               // acquire: don't read stale lines
}

// One 16x16 (MxN) bf16 WMMA tile over K range [k0,k1) with f32 accumulation.
// aptr: row-major [.., KDIM] bf16 (ushort), row already selected per lane (M = lane&15)
// bptr: row-major [.., KDIM] bf16 (ushort), row = output column  (N = lane&15)
__device__ __forceinline__ v8f wmma_span(const unsigned short* __restrict__ aptr,
                                         const unsigned short* __restrict__ bptr,
                                         int k0, int k1, int lhalf, v8f acc) {
  const int ka = lhalf * 8;    // A: lanes 16-31 take K+8..  chunks
  const int kb = lhalf * 8;    // B: lanes 16-31 take K+8..  chunks (same as A)
#pragma unroll 4
  for (int kk = k0; kk < k1; kk += 32) {
    v8bf alo = *(const v8bf*)(const void*)(aptr + kk + ka);
    v8bf ahi = *(const v8bf*)(const void*)(aptr + kk + ka + 16);
    v16bf a = __builtin_shufflevector(alo, ahi, 0,1,2,3,4,5,6,7,8,9,10,11,12,13,14,15);
    v8bf blo = *(const v8bf*)(const void*)(bptr + kk + kb);
    v8bf bhi = *(const v8bf*)(const void*)(bptr + kk + kb + 16);
    v16bf b = __builtin_shufflevector(blo, bhi, 0,1,2,3,4,5,6,7,8,9,10,11,12,13,14,15);
    acc = __builtin_amdgcn_wmma_f32_16x16x32_bf16(false, a, false, b, (short)0, acc,
                                                  false, false);
  }
  return acc;
}

// Convert weights to bf16 (Wz|Wr fused, Wc), seed h buffers from h0, reset barrier.
__global__ void __launch_bounds__(NTHR) gru_init(
    const float* __restrict__ Wz, const float* __restrict__ Wr,
    const float* __restrict__ Wc, const float* __restrict__ h0,
    unsigned short* __restrict__ Wzr_b, unsigned short* __restrict__ Wc_b,
    unsigned short* __restrict__ xh, float* __restrict__ hbuf,
    unsigned* __restrict__ arrive) {
  size_t idx0 = (size_t)blockIdx.x * blockDim.x + threadIdx.x;
  if (idx0 == 0) *arrive = 0u;    // fresh barrier state every launch (graph-replay safe)
  const size_t NW1 = (size_t)2048 * KDIM;
  const size_t NW2 = (size_t)1024 * KDIM;
  const size_t NH  = (size_t)BATCH * HID;
  const size_t total = NW1 + NW2 + NH;
  const size_t stride = (size_t)gridDim.x * blockDim.x;
  for (size_t i = idx0; i < total; i += stride) {
    if (i < NW1) {
      size_t row = i / KDIM, col = i - row * KDIM;
      float v = (row < 1024) ? Wz[row * KDIM + col] : Wr[(row - 1024) * KDIM + col];
      Wzr_b[i] = f2bfu(v);
    } else if (i < NW1 + NW2) {
      size_t j = i - NW1;
      Wc_b[j] = f2bfu(Wc[j]);
    } else {
      size_t j = i - NW1 - NW2;
      size_t b = j / HID, n = j - b * HID;
      float v = h0[j];
      hbuf[j] = v;
      xh[b * KDIM + DIN + n] = f2bfu(v);
    }
  }
}

__global__ void __launch_bounds__(NTHR, 1) gru_persistent(
    const float* __restrict__ x,
    const float* __restrict__ bz, const float* __restrict__ br,
    const float* __restrict__ bc,
    const unsigned short* __restrict__ Wzr,   // [2048][1536] bf16 (z rows then r rows)
    const unsigned short* __restrict__ Wc,    // [1024][1536] bf16
    unsigned short* __restrict__ xh,          // [64][1536] bf16: [x_t | h]
    unsigned short* __restrict__ xrh,         // [64][1536] bf16: [x_t | r*h]
    float* __restrict__ zbuf,                 // [64][1024]
    float* __restrict__ hbuf,                 // [64][1024]
    unsigned* __restrict__ arrive,
    float* __restrict__ out) {                // [512][64][1024]
  const int tid   = threadIdx.x;
  const int wg    = blockIdx.x;
  const int wave  = tid >> 5;
  const int lane  = tid & 31;
  const int lhalf = lane >> 4;
  const int l16   = lane & 15;

  __shared__ float redbuf[4 * 16 * 16];   // candidate-phase K-split reduction

  unsigned epoch = 0;

  for (int t = 0; t < TSTEPS; ++t) {
    // ---- phase A: bf16-convert x_t into xh[:, :512] and xrh[:, :512] ----
    {
      const float* xt = x + (size_t)t * (BATCH * DIN);
      for (int i = wg * NTHR + tid; i < BATCH * DIN; i += NWG * NTHR) {
        int b = i >> 9;          // /512
        int k = i & (DIN - 1);
        unsigned short v = f2bfu(xt[i]);
        xh[b * KDIM + k]  = v;
        xrh[b * KDIM + k] = v;
      }
    }
    ++epoch; grid_barrier(arrive, epoch * NWG);

    // ---- phase B: [64,2048] = xh @ [Wz;Wr]^T ; z -> zbuf, r -> r*h into xrh ----
    {
      const int tileid = wg * 8 + wave;       // 0..511
      const int m0 = (tileid & 3) * 16;       // batch tile
      const int n0 = (tileid >> 2) * 16;      // output column tile (0..2047)
      const unsigned short* aptr = xh  + (size_t)(m0 + l16) * KDIM;
      const unsigned short* bptr = Wzr + (size_t)(n0 + l16) * KDIM;
      v8f acc = {};
      acc = wmma_span(aptr, bptr, 0, KDIM, lhalf, acc);

      const bool isZ = (n0 < HID);
      const int col  = (isZ ? n0 : n0 - HID) + l16;   // 0..1023 within gate
      const float bias = isZ ? bz[col] : br[col];
#pragma unroll
      for (int r = 0; r < 8; ++r) {
        int b = m0 + r + lhalf * 8;                   // batch index
        float s = sigm(acc[r] + bias);
        if (isZ) {
          zbuf[b * HID + col] = s;
        } else {
          float rh = s * hbuf[b * HID + col];
          xrh[b * KDIM + DIN + col] = f2bfu(rh);
        }
      }
    }
    ++epoch; grid_barrier(arrive, epoch * NWG);

    // ---- phase C: hcand = tanh(xrh @ Wc^T + bc); h = (1-z)h + z*hcand ----
    {
      const int m0    = (wave & 3) * 16;
      const int khalf = wave >> 2;            // K split over wave pairs
      const int n0    = wg * 16;
      const unsigned short* aptr = xrh + (size_t)(m0 + l16) * KDIM;
      const unsigned short* bptr = Wc  + (size_t)(n0 + l16) * KDIM;
      v8f acc = {};
      acc = wmma_span(aptr, bptr, khalf * (KDIM / 2), (khalf + 1) * (KDIM / 2),
                      lhalf, acc);

      if (khalf == 1) {
#pragma unroll
        for (int r = 0; r < 8; ++r) {
          int rowin = r + lhalf * 8;
          redbuf[(wave & 3) * 256 + rowin * 16 + l16] = acc[r];
        }
      }
      __syncthreads();
      if (khalf == 0) {
        const int col = n0 + l16;
        const float bias = bc[col];
#pragma unroll
        for (int r = 0; r < 8; ++r) {
          int rowin = r + lhalf * 8;
          int b = m0 + rowin;
          float v = acc[r] + redbuf[(wave & 3) * 256 + rowin * 16 + l16] + bias;
          float hc = fast_tanh(v);
          float zv = zbuf[b * HID + col];
          float hp = hbuf[b * HID + col];
          float hn = (1.0f - zv) * hp + zv * hc;
          hbuf[b * HID + col] = hn;
          xh[b * KDIM + DIN + col] = f2bfu(hn);
          out[(size_t)t * (BATCH * HID) + b * HID + col] = hn;
        }
      }
    }
    ++epoch; grid_barrier(arrive, epoch * NWG);
  }
}

extern "C" void kernel_launch(void* const* d_in, const int* in_sizes, int n_in,
                              void* d_out, int out_size, void* d_ws, size_t ws_size,
                              hipStream_t stream) {
  (void)in_sizes; (void)n_in; (void)out_size; (void)ws_size;
  const float* x  = (const float*)d_in[0];
  const float* h0 = (const float*)d_in[1];
  const float* Wz = (const float*)d_in[2];
  const float* bz = (const float*)d_in[3];
  const float* Wr = (const float*)d_in[4];
  const float* br = (const float*)d_in[5];
  const float* Wc = (const float*)d_in[6];
  const float* bc = (const float*)d_in[7];
  float* out = (float*)d_out;

  char* ws = (char*)d_ws;
  size_t off = 0;
  unsigned* arrive      = (unsigned*)(ws + off);       off += 256;
  unsigned short* WzrB  = (unsigned short*)(ws + off); off += (size_t)2048 * KDIM * 2;
  unsigned short* WcB   = (unsigned short*)(ws + off); off += (size_t)1024 * KDIM * 2;
  unsigned short* xh    = (unsigned short*)(ws + off); off += (size_t)BATCH * KDIM * 2;
  unsigned short* xrh   = (unsigned short*)(ws + off); off += (size_t)BATCH * KDIM * 2;
  float* zbuf           = (float*)(ws + off);          off += (size_t)BATCH * HID * 4;
  float* hbuf           = (float*)(ws + off);          off += (size_t)BATCH * HID * 4;
  // total ~10.4 MB of workspace

  gru_init<<<2048, NTHR, 0, stream>>>(Wz, Wr, Wc, h0, WzrB, WcB, xh, hbuf, arrive);
  gru_persistent<<<NWG, NTHR, 0, stream>>>(x, bz, br, bc, WzrB, WcB,
                                           xh, xrh, zbuf, hbuf, arrive, out);
}
